// customDeformConv_53584011985325
// MI455X (gfx1250) — matrix operand order repaired and checked by reading, not compiled, not run on hardware
//
#include <hip/hip_runtime.h>
#include <hip/hip_bf16.h>
#include <stdint.h>

// ---- problem constants (match reference) ----
#define CI    64
#define CO    64
#define Hh    128
#define Ww    128
#define HWp   (Hh * Ww)          // 16384 = 1<<14
#define Bb    8
#define K2c   9
#define KDIM  (CI * K2c)         // 576 (reduction dim, k = c*9 + k2)
#define A_STRIDE (KDIM + 8)      // padded bf16 row stride (avoids LDS bank clash)
#define NOM   32                 // offset+mask channels padded: 18+9 -> 32

typedef __attribute__((ext_vector_type(16))) __bf16 v16bf;
typedef __attribute__((ext_vector_type(8)))  float  v8f;

__device__ __forceinline__ unsigned short f2bf(float f) {
  unsigned u = __float_as_uint(f);
  u += 0x7FFFu + ((u >> 16) & 1u);   // round-to-nearest-even
  return (unsigned short)(u >> 16);
}

// ---------------------------------------------------------------------------
// Kernel 0a: repack main conv weight [Co][Ci][3][3] f32 -> bf16 [Co][576]
// (flat layout already matches k = c*9 + (ky*3+kx))
// ---------------------------------------------------------------------------
__global__ void pack_w_kernel(const float* __restrict__ w,
                              unsigned short* __restrict__ wp) {
  int i = blockIdx.x * blockDim.x + threadIdx.x;
  if (i < CO * KDIM) wp[i] = f2bf(w[i]);
}

// ---------------------------------------------------------------------------
// Kernel 0b: pack offset+mask weights -> bf16 [32][576]
// rows 0-17 = off_w, rows 18-26 = mask_w, rows 27-31 = zero padding
// ---------------------------------------------------------------------------
__global__ void pack_offmask_w_kernel(const float* __restrict__ off_w,
                                      const float* __restrict__ mask_w,
                                      unsigned short* __restrict__ wp2) {
  int i = blockIdx.x * blockDim.x + threadIdx.x;   // flat [32*576)
  if (i >= NOM * KDIM) return;
  float v = 0.0f;
  if (i < 18 * KDIM)      v = off_w[i];
  else if (i < 27 * KDIM) v = mask_w[i - 18 * KDIM];
  wp2[i] = f2bf(v);
}

// ---------------------------------------------------------------------------
// Kernel 1: offset (18ch) + mask (9ch, sigmoid) 3x3 convs as im2col WMMA GEMM.
// WG = 256 thr = 8 waves, 32 pixels x 32 channels, K = 576 split across
// wave pairs (kh) with LDS partial reduction. 9 WMMAs per wave.
// ---------------------------------------------------------------------------
__global__ __launch_bounds__(256) void offmask_wmma_kernel(
    const float* __restrict__ x, const unsigned short* __restrict__ wp2,
    const float* __restrict__ off_b, const float* __restrict__ mask_b,
    float* __restrict__ off_out, float* __restrict__ mask_out) {
  __shared__ unsigned short A_lds[32 * A_STRIDE];   // 37376 B
  __shared__ float partial[4 * 16 * 17];            //  4352 B (padded rows)

  int blk = blockIdx.x;
  int b   = blk >> 9;                 // 512 blocks per batch
  int pxb = (blk & 511) << 5;         // 32-pixel tile (within one image row)
  int tid = threadIdx.x;

  // ---- stage im2col patch (zero-padded taps) as bf16 A tile ----
  const float* xb = x + ((size_t)b << 20);
  for (int it = 0; it < (32 * KDIM) / 256; ++it) {   // 72 iters
    int s  = it * 256 + tid;
    int ip = s & 31;            // pixel in tile (lanes = adjacent pixels)
    int k  = s >> 5;            // k = c*9 + k2, uniform per wave
    int c  = k / 9, k2 = k - c * 9;
    int p  = pxb + ip;
    int ho = p >> 7, wo = p & 127;
    int y  = ho - 1 + k2 / 3;
    int xx = wo - 1 + k2 % 3;
    bool v = (y >= 0) && (y < Hh) && (xx >= 0) && (xx < Ww);
    float val = v ? xb[(c << 14) + (y << 7) + xx] : 0.0f;
    A_lds[ip * A_STRIDE + k] = f2bf(val);
  }
  __syncthreads();

  int wave = tid >> 5;
  int lane = tid & 31;
  int mt = wave & 1;            // pixel tile (2 x 16)
  int nt = (wave >> 1) & 1;     // channel tile (2 x 16)
  int kh = wave >> 2;           // K half: 0 -> K[0,288), 1 -> K[288,576)
  int lm = lane & 15;
  int hi = lane >> 4;

  v8f acc = {};
  const unsigned short* A_row =
      &A_lds[(mt * 16 + lm) * A_STRIDE + kh * (KDIM / 2)];
  const unsigned short* B_row =
      wp2 + (size_t)(nt * 16 + lm) * KDIM + kh * (KDIM / 2);
  for (int kk = 0; kk < KDIM / 64; ++kk) {          // 9 iters per K half
    int ka = kk * 32 + hi * 8;
    union { uint4 u[2]; v16bf v; } ua, ub;
    ua.u[0] = *(const uint4*)(A_row + ka);          // ds_read_b128
    ua.u[1] = *(const uint4*)(A_row + ka + 16);
    int kb = kk * 32 + hi * 8;
    ub.u[0] = *(const uint4*)(B_row + kb);          // global_load_b128
    ub.u[1] = *(const uint4*)(B_row + kb + 16);
    acc = __builtin_amdgcn_wmma_f32_16x16x32_bf16(
        false, ua.v, false, ub.v, (short)0, acc, false, false);
  }

  // ---- K-split reduction through LDS (pairs: wave w <- wave w+4) ----
  float* pt = &partial[(wave & 3) * 16 * 17];
  if (kh == 1) {
#pragma unroll
    for (int j = 0; j < 8; ++j) pt[(j + 8 * hi) * 17 + lm] = acc[j];
  }
  __syncthreads();
  if (kh == 0) {
#pragma unroll
    for (int j = 0; j < 8; ++j) acc[j] += pt[(j + 8 * hi) * 17 + lm];

    int ch = nt * 16 + lm;      // 0..31 (27..31 = padding, dropped)
    if (ch < 27) {
      float bv = (ch < 18) ? off_b[ch] : mask_b[ch - 18];
#pragma unroll
      for (int r = 0; r < 8; ++r) {
        int p = pxb + mt * 16 + r + 8 * hi;
        float v = acc[r] + bv;
        if (ch < 18)
          off_out[(((size_t)(b * 18 + ch)) << 14) + p] = v;
        else
          mask_out[(((size_t)(b * 9 + (ch - 18))) << 14) + p] =
              1.0f / (1.0f + __expf(-v));
      }
    }
  }
}

// ---------------------------------------------------------------------------
// Kernel 2: fused bilinear sampling + WMMA GEMM (unchanged from round 1).
// WG = 256 thr = 8 waves, 32 pixels x 64 co of one batch.
//  phase A: per-(px,k2) bilinear meta (4 clamped indices + 4 mask-scaled wts)
//  phase B: gather/sample 32x576 tile -> bf16 A in LDS (each sample computed
//           once, shared across all 64 output channels)
//  phase C: each wave: one 16x16 (M,N) tile, 18 x v_wmma_f32_16x16x32_bf16
// ---------------------------------------------------------------------------
__global__ __launch_bounds__(256) void deform_gemm_kernel(
    const float* __restrict__ x, const unsigned short* __restrict__ wp,
    const float* __restrict__ bias,
    const float* __restrict__ off, const float* __restrict__ msk,
    float* __restrict__ out) {
  __shared__ unsigned short A_lds[32 * A_STRIDE];   // 37376 B
  __shared__ int4   midx[32 * K2c];                 //  4608 B
  __shared__ float4 mwgt[32 * K2c];                 //  4608 B

  int blk = blockIdx.x;
  int b   = blk >> 9;
  int pxb = (blk & 511) << 5;
  int tid = threadIdx.x;

  // ---- phase A: bilinear meta per (pixel-in-tile, tap) ----
  for (int e = tid; e < 32 * K2c; e += 256) {
    int ip = e & 31, k2 = e >> 5;
    int p  = pxb + ip;
    int ho = p >> 7, wo = p & 127;
    float dy = off[(((size_t)(b * 18 + 2 * k2)) << 14) + p];
    float dx = off[(((size_t)(b * 18 + 2 * k2 + 1)) << 14) + p];
    float mm = msk[(((size_t)(b * 9 + k2)) << 14) + p];
    float py = (float)(ho - 1 + k2 / 3) + dy;
    float px = (float)(wo - 1 + k2 % 3) + dx;
    float y0f = floorf(py), x0f = floorf(px);
    float ly = py - y0f,   lx = px - x0f;
    int y0 = (int)y0f, x0 = (int)x0f;
    int y1 = y0 + 1,   x1 = x0 + 1;
    bool vy0 = (y0 >= 0) && (y0 < Hh), vy1 = (y1 >= 0) && (y1 < Hh);
    bool vx0 = (x0 >= 0) && (x0 < Ww), vx1 = (x1 >= 0) && (x1 < Ww);
    int yc0 = min(max(y0, 0), Hh - 1), yc1 = min(max(y1, 0), Hh - 1);
    int xc0 = min(max(x0, 0), Ww - 1), xc1 = min(max(x1, 0), Ww - 1);
    float w00 = (vy0 && vx0) ? (1.0f - ly) * (1.0f - lx) * mm : 0.0f;
    float w01 = (vy0 && vx1) ? (1.0f - ly) * lx * mm : 0.0f;
    float w10 = (vy1 && vx0) ? ly * (1.0f - lx) * mm : 0.0f;
    float w11 = (vy1 && vx1) ? ly * lx * mm : 0.0f;
    midx[e] = make_int4((yc0 << 7) + xc0, (yc0 << 7) + xc1,
                        (yc1 << 7) + xc0, (yc1 << 7) + xc1);
    mwgt[e] = make_float4(w00, w01, w10, w11);
  }
  __syncthreads();

  // ---- phase B: sample 32x576 tile into bf16 A (lanes = adjacent pixels) ----
  const float* xb = x + ((size_t)b << 20);
  for (int it = 0; it < (32 * KDIM) / 256; ++it) {   // 72 iters
    int s  = it * 256 + tid;
    int ip = s & 31;
    int k  = s >> 5;
    int c  = k / 9, k2 = k - c * 9;
    int e  = (k2 << 5) + ip;
    int4   id = midx[e];
    float4 w  = mwgt[e];
    const float* xc = xb + (c << 14);
    float v = w.x * xc[id.x] + w.y * xc[id.y] + w.z * xc[id.z] + w.w * xc[id.w];
    A_lds[ip * A_STRIDE + k] = f2bf(v);
  }
  __syncthreads();

  // ---- phase C: WMMA GEMM, D[16 px][16 co] per wave, K = 576 ----
  int wave = tid >> 5;
  int lane = tid & 31;
  int mt = wave & 1;            // pixel tile (2 x 16)
  int nt = wave >> 1;           // co tile    (4 x 16)
  int lm = lane & 15;
  int hi = lane >> 4;

  v8f acc = {};
  const unsigned short* A_row = &A_lds[(mt * 16 + lm) * A_STRIDE];
  const unsigned short* B_row = wp + (size_t)(nt * 16 + lm) * KDIM;
  for (int kk = 0; kk < KDIM / 32; ++kk) {          // 18 iters
    int ka = kk * 32 + hi * 8;
    union { uint4 u[2]; v16bf v; } ua, ub;
    ua.u[0] = *(const uint4*)(A_row + ka);          // ds_read_b128
    ua.u[1] = *(const uint4*)(A_row + ka + 16);
    int kb = kk * 32 + hi * 8;
    ub.u[0] = *(const uint4*)(B_row + kb);          // global_load_b128
    ub.u[1] = *(const uint4*)(B_row + kb + 16);
    acc = __builtin_amdgcn_wmma_f32_16x16x32_bf16(
        false, ua.v, false, ub.v, (short)0, acc, false, false);
  }

  // ---- epilogue: C/D layout (VGPR j -> M=j / j+8; N = lane%16) + bias ----
  int co = nt * 16 + lm;
  float bv = bias[co];
  float* ob = out + (((size_t)(b * CO + co)) << 14);
#pragma unroll
  for (int j = 0; j < 8; ++j) {
    int m = j + hi * 8;
    ob[pxb + mt * 16 + m] = acc[j] + bv;
  }
}

// ---------------------------------------------------------------------------
extern "C" void kernel_launch(void* const* d_in, const int* in_sizes, int n_in,
                              void* d_out, int out_size, void* d_ws, size_t ws_size,
                              hipStream_t stream) {
  const float* x      = (const float*)d_in[0];
  const float* weight = (const float*)d_in[1];
  const float* bias   = (const float*)d_in[2];
  const float* off_w  = (const float*)d_in[3];
  const float* off_b  = (const float*)d_in[4];
  const float* mask_w = (const float*)d_in[5];
  const float* mask_b = (const float*)d_in[6];
  float* out = (float*)d_out;

  // workspace: offset [B,18,HW] f32 | mask [B,9,HW] f32 | wpk bf16 | wpk2 bf16
  float* off = (float*)d_ws;
  float* msk = off + (size_t)Bb * 18 * HWp;
  unsigned short* wpk  = (unsigned short*)(msk + (size_t)Bb * 9 * HWp);
  unsigned short* wpk2 = wpk + (size_t)CO * KDIM;

  pack_w_kernel<<<(CO * KDIM + 255) / 256, 256, 0, stream>>>(weight, wpk);
  pack_offmask_w_kernel<<<(NOM * KDIM + 255) / 256, 256, 0, stream>>>(
      off_w, mask_w, wpk2);
  offmask_wmma_kernel<<<Bb * 512, 256, 0, stream>>>(
      x, wpk2, off_b, mask_b, off, msk);
  deform_gemm_kernel<<<Bb * 512, 256, 0, stream>>>(
      x, wpk, bias, off, msk, out);
}
